// RoPE_mha_15401752723834
// MI455X (gfx1250) — matrix operand order repaired and checked by reading, not compiled, not run on hardware
//
#include <hip/hip_runtime.h>
#include <hip/hip_bf16.h>

typedef __attribute__((ext_vector_type(16))) __bf16 v16bf;
typedef __attribute__((ext_vector_type(8)))  __bf16 v8bf;
typedef __attribute__((ext_vector_type(8)))  float  v8f;

#define D_MODEL  2048
#define NHEAD    16
#define HEAD_DIM 128
#define BATCH    2
#define SEQ      2048
#define MTOT     (BATCH * SEQ)   // 4096

// ---------------------------------------------------------------------------
// CDNA5 async copy helpers
// ---------------------------------------------------------------------------
__device__ __forceinline__ void async_copy_b128(const __bf16* g, unsigned lds_off) {
    unsigned long long ga = (unsigned long long)g;
    // GLOBAL_LOAD_ASYNC_TO_LDS_B128: VDST = LDS byte addr VGPR, VADDR = 64-bit global addr
    asm volatile("global_load_async_to_lds_b128 %0, %1, off"
                 :: "v"(lds_off), "v"(ga) : "memory");
}
__device__ __forceinline__ void wait_asynccnt0() {
    asm volatile("s_wait_asynccnt 0x0" ::: "memory");
}

#if __has_builtin(__builtin_amdgcn_tensor_load_to_lds)
#define HAVE_TDM 1
typedef unsigned int v4u __attribute__((ext_vector_type(4)));
typedef int          v8i __attribute__((ext_vector_type(8)));
typedef int          v4i __attribute__((ext_vector_type(4)));
// 2-D TDM load: rows x row_elems bf16 tile, row stride in elements (D# per ISA 8.3/8.4)
__device__ __forceinline__ void tdm_load_2d(const __bf16* gsrc, unsigned lds_off,
                                            int rows, int row_elems, int row_stride) {
    unsigned long long ga = (unsigned long long)gsrc;
    v4u g0;
    g0[0] = 1u;                                       // count=1, user descriptor
    g0[1] = lds_off;                                  // lds_addr (bytes)
    g0[2] = (unsigned)ga;                             // global_addr[31:0]
    g0[3] = (unsigned)(ga >> 32) | (2u << 30);        // global_addr[56:32] | type=2
    v8i g1;
    g1[0] = (1 << 16);                                // wg_mask=0, data_size=1 (2 bytes)
    g1[1] = (row_stride & 0xFFFF) << 16;              // tensor_dim0[15:0]
    g1[2] = ((row_stride >> 16) & 0xFFFF)             // tensor_dim0[31:16]
          | ((rows & 0xFFFF) << 16);                  // tensor_dim1[15:0]
    g1[3] = (row_elems & 0xFFFF) << 16;               // tensor_dim1[31:16]=0, tile_dim0
    g1[4] = rows & 0xFFFF;                            // tile_dim1, tile_dim2=0
    g1[5] = row_stride;                               // tensor_dim0_stride[31:0]
    g1[6] = 0;                                        // stride[47:32], dim1_stride lo
    g1[7] = 0;
    v4i z4 = {0, 0, 0, 0};
    v8i z8 = {0, 0, 0, 0, 0, 0, 0, 0};
    __builtin_amdgcn_tensor_load_to_lds(g0, g1, z4, z4, z8, 0);
}
#endif

// ---------------------------------------------------------------------------
// fp32 -> bf16 cast
// ---------------------------------------------------------------------------
__global__ void cvt_f32_bf16(const float* __restrict__ in, __bf16* __restrict__ out, int n) {
    int i = blockIdx.x * blockDim.x + threadIdx.x;
    if (i < n) out[i] = (__bf16)in[i];
}

// ---------------------------------------------------------------------------
// C[M,N] = A[M,K] (bf16) @ W[N,K]^T (bf16) + bias(f32)
// mode 0: out f32 [M,N]
// mode 1: out bf16 remapped to [B,H,S,HEAD_DIM]   (Q/K projections)
// mode 2: out bf16 remapped to [B,H,HEAD_DIM,S]   (V projection, transposed)
// Block 256 thr = 8 waves (4m x 2n); block tile 128x128; wave tile 32x64.
// Double-buffered LDS; A via ASYNC loads, W via TDM; one barrier per K-step.
// ---------------------------------------------------------------------------
__launch_bounds__(256)
__global__ void gemm_wmma(const __bf16* __restrict__ A, const __bf16* __restrict__ W,
                          const float* __restrict__ bias, void* __restrict__ out,
                          int M, int N, int K, int mode)
{
    __shared__ alignas(64) __bf16 As[2][128][32];
    __shared__ alignas(64) __bf16 Ws[2][128][32];

    const int tid  = threadIdx.x;
    const int lane = tid & 31;
    const int wid  = tid >> 5;
    const int wm   = wid >> 1;     // 0..3
    const int wn   = wid & 1;      // 0..1
    const int half = lane >> 4;    // 0/1
    const int l15  = lane & 15;
    const int m0   = blockIdx.y * 128;
    const int n0   = blockIdx.x * 128;

    const int sr = tid >> 1;            // staging row 0..127
    const int sc = (tid & 1) * 16;      // staging col (elements)

    v8f acc[2][4] = {};

    // prologue: issue fills for K-chunk 0 into buffer 0
    {
        const __bf16* ga = &A[(size_t)(m0 + sr) * K + sc];
        unsigned lo = (unsigned)(uintptr_t)&As[0][sr][sc];
        async_copy_b128(ga,     lo);
        async_copy_b128(ga + 8, lo + 16);
#if HAVE_TDM
        if (wid == 0)
            tdm_load_2d(&W[(size_t)n0 * K], (unsigned)(uintptr_t)&Ws[0][0][0], 128, 32, K);
#else
        const __bf16* gw = &W[(size_t)(n0 + sr) * K + sc];
        unsigned lw = (unsigned)(uintptr_t)&Ws[0][sr][sc];
        async_copy_b128(gw,     lw);
        async_copy_b128(gw + 8, lw + 16);
#endif
    }

    int cur = 0;
    for (int k0 = 0; k0 < K; k0 += 32, cur ^= 1) {
        // wait for buf[cur] fills (own wave's async + TDM), then block-wide sync
        wait_asynccnt0();
#if HAVE_TDM
        if (wid == 0) __builtin_amdgcn_s_wait_tensorcnt(0);
#endif
        __syncthreads();

        // issue fills for next K-chunk into buf[cur^1] (safe: one barrier back,
        // every wave has finished reading buf[cur^1] from the previous step)
        if (k0 + 32 < K) {
            const __bf16* ga = &A[(size_t)(m0 + sr) * K + (k0 + 32) + sc];
            unsigned lo = (unsigned)(uintptr_t)&As[cur ^ 1][sr][sc];
            async_copy_b128(ga,     lo);
            async_copy_b128(ga + 8, lo + 16);
#if HAVE_TDM
            if (wid == 0)
                tdm_load_2d(&W[(size_t)n0 * K + (k0 + 32)],
                            (unsigned)(uintptr_t)&Ws[cur ^ 1][0][0], 128, 32, K);
#else
            const __bf16* gw = &W[(size_t)(n0 + sr) * K + (k0 + 32) + sc];
            unsigned lw = (unsigned)(uintptr_t)&Ws[cur ^ 1][sr][sc];
            async_copy_b128(gw,     lw);
            async_copy_b128(gw + 8, lw + 16);
#endif
        }

        // fragments from buf[cur] + 8 WMMAs per wave
        v16bf af[2], bw[4];
        const int koff = half * 16;
#pragma unroll
        for (int i = 0; i < 2; ++i) {
            int mr = wm * 32 + i * 16 + l15;
            ((v8bf*)&af[i])[0] = *(const v8bf*)&As[cur][mr][koff];
            ((v8bf*)&af[i])[1] = *(const v8bf*)&As[cur][mr][koff + 8];
        }
#pragma unroll
        for (int j = 0; j < 4; ++j) {
            int nr = wn * 64 + j * 16 + l15;
            bw[j] = *(const v16bf*)&Ws[cur][nr][half * 16];
        }
#pragma unroll
        for (int i = 0; i < 2; ++i)
#pragma unroll
            for (int j = 0; j < 4; ++j)
                acc[i][j] = __builtin_amdgcn_wmma_f32_16x16x32_bf16(
                    false, af[i], false, bw[j], (short)0, acc[i][j], false, false);
    }

    // epilogue
#pragma unroll
    for (int i = 0; i < 2; ++i)
#pragma unroll
        for (int j = 0; j < 4; ++j)
#pragma unroll
            for (int r = 0; r < 8; ++r) {
                int mg = m0 + wm * 32 + i * 16 + r + half * 8;
                int ng = n0 + wn * 64 + j * 16 + l15;
                float v = acc[i][j][r] + bias[ng];
                if (mode == 0) {
                    ((float*)out)[(size_t)mg * N + ng] = v;
                } else {
                    int b = mg >> 11, s = mg & (SEQ - 1);
                    int h = ng >> 7, d = ng & (HEAD_DIM - 1);
                    __bf16* o = (__bf16*)out;
                    if (mode == 1)
                        o[(((size_t)(b * NHEAD + h)) * SEQ + s) * HEAD_DIM + d] = (__bf16)v;
                    else
                        o[(((size_t)(b * NHEAD + h)) * HEAD_DIM + d) * SEQ + s] = (__bf16)v;
                }
            }
}

// ---------------------------------------------------------------------------
// In-place RoPE on bf16 tensor [B,H,S,HEAD_DIM]; rotate first 64, quarter 32.
// ---------------------------------------------------------------------------
__global__ void rope_kernel(__bf16* __restrict__ X) {
    int idx = blockIdx.x * blockDim.x + threadIdx.x;
    if (idx >= BATCH * NHEAD * SEQ * 32) return;
    int d  = idx & 31;
    int s  = (idx >> 5) & (SEQ - 1);
    int bh = idx >> 16;
    __bf16* p = X + ((size_t)bh * SEQ + s) * HEAD_DIM;
    float x1 = (float)p[d];
    float x2 = (float)p[d + 32];
    const float LOG_T = 9.210340371976184f;   // ln(10000)
    float inv1 = __expf(-((2.0f * d) / 128.0f) * LOG_T);
    float inv2 = __expf(-((2.0f * (d + 32)) / 128.0f) * LOG_T);
    float a1 = (float)s * inv1, a2 = (float)s * inv2;
    float o1 = x1 * __cosf(a1) - x2 * __sinf(a1);
    float o2 = x2 * __cosf(a2) + x1 * __sinf(a2);
    p[d]      = (__bf16)o1;
    p[d + 32] = (__bf16)o2;
}

// ---------------------------------------------------------------------------
// Causal flash attention, bf16 WMMA, f32 online softmax.
// Q,K: [B,H,S,128] bf16;  Vt: [B,H,128,S] bf16;  O: [B,S,D_MODEL] bf16.
// ---------------------------------------------------------------------------
__launch_bounds__(128)
__global__ void attn_wmma(const __bf16* __restrict__ Q, const __bf16* __restrict__ Km,
                          const __bf16* __restrict__ Vt, __bf16* __restrict__ O)
{
    __shared__ alignas(64) __bf16 Pl[4][16][32];   // per-wave P staging (C->A relayout)

    const int tid  = threadIdx.x;
    const int lane = tid & 31;
    const int wid  = tid >> 5;
    const int half = lane >> 4;
    const int l15  = lane & 15;
    const int bh   = blockIdx.x;                   // b*NHEAD + h
    const int qt   = blockIdx.y;                   // q tile of 64 rows
    const int q0   = qt * 64 + wid * 16;

    const __bf16* qb = Q  + (size_t)bh * SEQ * HEAD_DIM;
    const __bf16* kb = Km + (size_t)bh * SEQ * HEAD_DIM;
    const __bf16* vb = Vt + (size_t)bh * HEAD_DIM * SEQ;

    v16bf qf[4];
    {
        const __bf16* qrow = qb + (size_t)(q0 + l15) * HEAD_DIM;
        const int koff = half * 16;
#pragma unroll
        for (int dc = 0; dc < 4; ++dc) {
            ((v8bf*)&qf[dc])[0] = *(const v8bf*)&qrow[dc * 32 + koff];
            ((v8bf*)&qf[dc])[1] = *(const v8bf*)&qrow[dc * 32 + koff + 8];
        }
    }

    v8f oacc[8] = {};
    float mrow[8], lrow[8];
#pragma unroll
    for (int r = 0; r < 8; ++r) { mrow[r] = -3.0e38f; lrow[r] = 0.0f; }
    const float scale = 0.08838834764831845f;      // 1/sqrt(128)
    const int nkt = 2 * (qt + 1);

    for (int kt = 0; kt < nkt; ++kt) {
        const int key0 = kt * 32;
        v8f s0 = {}, s1 = {};
#pragma unroll
        for (int dc = 0; dc < 4; ++dc) {
            const __bf16* kp = kb + (size_t)(key0 + l15) * HEAD_DIM + dc * 32 + half * 16;
            v16bf b0 = *(const v16bf*)kp;
            v16bf b1 = *(const v16bf*)(kp + 16 * HEAD_DIM);
            s0 = __builtin_amdgcn_wmma_f32_16x16x32_bf16(false, qf[dc], false, b0, (short)0, s0, false, false);
            s1 = __builtin_amdgcn_wmma_f32_16x16x32_bf16(false, qf[dc], false, b1, (short)0, s1, false, false);
        }

        float p0[8], p1[8];
#pragma unroll
        for (int r = 0; r < 8; ++r) {
            const int qg  = q0 + half * 8 + r;
            const int kg0 = key0 + l15;
            const int kg1 = key0 + 16 + l15;
            float v0 = (kg0 <= qg) ? s0[r] * scale : -3.0e38f;
            float v1 = (kg1 <= qg) ? s1[r] * scale : -3.0e38f;
            float mx = fmaxf(v0, v1);
#pragma unroll
            for (int sh = 8; sh >= 1; sh >>= 1) mx = fmaxf(mx, __shfl_xor(mx, sh, 32));
            float mnew = fmaxf(mrow[r], mx);
            float e0 = (kg0 <= qg) ? __expf(v0 - mnew) : 0.0f;
            float e1 = (kg1 <= qg) ? __expf(v1 - mnew) : 0.0f;
            float sum = e0 + e1;
#pragma unroll
            for (int sh = 8; sh >= 1; sh >>= 1) sum += __shfl_xor(sum, sh, 32);
            float alpha = __expf(mrow[r] - mnew);
            lrow[r] = lrow[r] * alpha + sum;
            mrow[r] = mnew;
#pragma unroll
            for (int dc2 = 0; dc2 < 8; ++dc2) oacc[dc2][r] *= alpha;
            p0[r] = e0;
            p1[r] = e1;
        }

#pragma unroll
        for (int r = 0; r < 8; ++r) {
            Pl[wid][half * 8 + r][l15]      = (__bf16)p0[r];
            Pl[wid][half * 8 + r][16 + l15] = (__bf16)p1[r];
        }
        v16bf ap;
        {
            const int koff = half * 16;
            ((v8bf*)&ap)[0] = *(const v8bf*)&Pl[wid][l15][koff];
            ((v8bf*)&ap)[1] = *(const v8bf*)&Pl[wid][l15][koff + 8];
        }

#pragma unroll
        for (int dc2 = 0; dc2 < 8; ++dc2) {
            const __bf16* vp = vb + (size_t)(dc2 * 16 + l15) * SEQ + key0 + half * 16;
            v16bf bv = *(const v16bf*)vp;
            oacc[dc2] = __builtin_amdgcn_wmma_f32_16x16x32_bf16(
                false, ap, false, bv, (short)0, oacc[dc2], false, false);
        }
    }

    const int b = bh >> 4, h = bh & (NHEAD - 1);
#pragma unroll
    for (int dc2 = 0; dc2 < 8; ++dc2)
#pragma unroll
        for (int r = 0; r < 8; ++r) {
            int s = q0 + half * 8 + r;
            int d = dc2 * 16 + l15;
            float v = oacc[dc2][r] / lrow[r];
            O[((size_t)(b * SEQ + s)) * D_MODEL + h * HEAD_DIM + d] = (__bf16)v;
        }
}

// ---------------------------------------------------------------------------
extern "C" void kernel_launch(void* const* d_in, const int* in_sizes, int n_in,
                              void* d_out, int out_size, void* d_ws, size_t ws_size,
                              hipStream_t stream) {
    (void)in_sizes; (void)n_in; (void)out_size; (void)ws_size;
    const float* x  = (const float*)d_in[0];
    // d_in[1] = causal mask, handled analytically
    const float* Wq = (const float*)d_in[2];
    const float* bq = (const float*)d_in[3];
    const float* Wk = (const float*)d_in[4];
    const float* bk = (const float*)d_in[5];
    const float* Wv = (const float*)d_in[6];
    const float* bv = (const float*)d_in[7];
    const float* Wo = (const float*)d_in[8];
    const float* bo = (const float*)d_in[9];

    const size_t NX = (size_t)MTOT * D_MODEL;      // 8M elems
    const size_t NW = (size_t)D_MODEL * D_MODEL;   // 4M elems
    __bf16* xb   = (__bf16*)d_ws;
    __bf16* wqb  = xb  + NX;
    __bf16* wkb  = wqb + NW;
    __bf16* wvb  = wkb + NW;
    __bf16* wob  = wvb + NW;
    __bf16* qb   = wob + NW;
    __bf16* kbuf = qb  + NX;
    __bf16* vtb  = kbuf + NX;
    __bf16* ob   = vtb + NX;   // ~112 MB bf16 workspace

    cvt_f32_bf16<<<(int)((NX + 255) / 256), 256, 0, stream>>>(x,  xb,  (int)NX);
    cvt_f32_bf16<<<(int)((NW + 255) / 256), 256, 0, stream>>>(Wq, wqb, (int)NW);
    cvt_f32_bf16<<<(int)((NW + 255) / 256), 256, 0, stream>>>(Wk, wkb, (int)NW);
    cvt_f32_bf16<<<(int)((NW + 255) / 256), 256, 0, stream>>>(Wv, wvb, (int)NW);
    cvt_f32_bf16<<<(int)((NW + 255) / 256), 256, 0, stream>>>(Wo, wob, (int)NW);

    dim3 gg(D_MODEL / 128, MTOT / 128);
    gemm_wmma<<<gg, 256, 0, stream>>>(xb, wqb, bq, qb,   MTOT, D_MODEL, D_MODEL, 1);
    gemm_wmma<<<gg, 256, 0, stream>>>(xb, wkb, bk, kbuf, MTOT, D_MODEL, D_MODEL, 1);
    gemm_wmma<<<gg, 256, 0, stream>>>(xb, wvb, bv, vtb,  MTOT, D_MODEL, D_MODEL, 2);

    const int nrope = BATCH * NHEAD * SEQ * 32;
    rope_kernel<<<nrope / 256, 256, 0, stream>>>(qb);
    rope_kernel<<<nrope / 256, 256, 0, stream>>>(kbuf);

    dim3 ga(BATCH * NHEAD, SEQ / 64);
    attn_wmma<<<ga, 128, 0, stream>>>(qb, kbuf, vtb, ob);

    gemm_wmma<<<gg, 256, 0, stream>>>(ob, wob, bo, d_out, MTOT, D_MODEL, D_MODEL, 0);
}
